// AttnDecoder_54030688583864
// MI455X (gfx1250) — matrix operand order repaired and checked by reading, not compiled, not run on hardware
//
#include <hip/hip_runtime.h>
#include <hip/hip_bf16.h>

#define BB 32
#define SS 128
#define HH 1024
#define VV 32000
#define TT 50
#define LL 3
#define BH (BB * HH)          // 32768
#define BSH (BB * SS * HH)    // 4194304

typedef __attribute__((ext_vector_type(16))) __bf16 v16bf;
typedef __attribute__((ext_vector_type(8)))  float  v8f;

union FragU { v16bf v; uint4 q[2]; };

__device__ __forceinline__ unsigned short f2bf(float f) {
  union { float f; unsigned u; } x; x.f = f;
  unsigned u = x.u;
  unsigned r = (u + 0x7fffu + ((u >> 16) & 1u)) >> 16;   // round-to-nearest-even
  return (unsigned short)r;
}

__device__ __forceinline__ float sigf(float x) { return 1.f / (1.f + __expf(-x)); }

// Block-wide sum reduction, blockDim.x == 256, red is shared float[256].
__device__ __forceinline__ float block_sum(float v, float* red) {
  int tid = threadIdx.x;
  red[tid] = v; __syncthreads();
  for (int o = 128; o > 0; o >>= 1) { if (tid < o) red[tid] += red[tid + o]; __syncthreads(); }
  float r = red[0]; __syncthreads();
  return r;
}

// ---------------- fragment loads ----------------
// A matrix (16x32 bf16, row-major source, row stride K ushorts, K % 32 == 0):
// lane: m = m0 + (lane&15), kb = (lane>>4)*8; elems 0..7 -> k=kb+e, 8..15 -> k=kb+16+(e-8)
__device__ __forceinline__ v16bf load_afrag(const unsigned short* __restrict__ A,
                                            int m0, int K, int k0, int lane) {
  int m  = m0 + (lane & 15);
  int kb = (lane >> 4) << 3;
  const unsigned short* p = A + (size_t)m * K + k0 + kb;
  FragU f;
  f.q[0] = *(const uint4*)(p);        // k = kb + 0..7
  f.q[1] = *(const uint4*)(p + 16);   // k = kb + 16..23
  return f.v;
}

// B matrix fragment: pre-packed fragment-major, 512 ushorts per 16(N)x32(K) tile.
__device__ __forceinline__ v16bf load_bfrag(const unsigned short* __restrict__ Wp,
                                            size_t tile, int lane) {
  const unsigned short* p = Wp + tile * 512 + (size_t)lane * 16;
  FragU f;
  f.q[0] = *(const uint4*)(p);
  f.q[1] = *(const uint4*)(p + 8);
  return f.v;
}

// ---------------- prep kernels ----------------
// Pack W (N x K fp32, row-major) into bf16 WMMA-B fragment-major layout.
// B-layout per tile: lane n = lane&15; k = (lane>>4)*8 + e for e<8, (lane>>4)*8 + 16 + (e-8) for e>=8
__global__ void k_pack_w(const float* __restrict__ W, unsigned short* __restrict__ out,
                         int N, int K) {
  size_t total = (size_t)N * K;
  int kt = K >> 5;
  for (size_t o = (size_t)blockIdx.x * blockDim.x + threadIdx.x; o < total;
       o += (size_t)gridDim.x * blockDim.x) {
    int e    = (int)(o & 15);
    int lane = (int)((o >> 4) & 31);
    size_t tile = o >> 9;
    int n  = (int)(tile / kt) * 16 + (lane & 15);
    int k0 = (int)(tile % kt) << 5;
    int k  = k0 + ((lane >> 4) << 3) + (e & 7) + ((e >> 3) << 4);
    out[o] = f2bf(W[(size_t)n * K + k]);
  }
}

__global__ void k_f2bf(const float* __restrict__ in, unsigned short* __restrict__ out, size_t n) {
  for (size_t i = (size_t)blockIdx.x * blockDim.x + threadIdx.x; i < n;
       i += (size_t)gridDim.x * blockDim.x)
    out[i] = f2bf(in[i]);
}

__global__ void k_init_state(const float* __restrict__ h0, const float* __restrict__ c0,
                             float* __restrict__ h, float* __restrict__ c,
                             unsigned short* __restrict__ hb) {
  int i = blockIdx.x * 256 + threadIdx.x;   // grid covers LL*BH exactly
  h[i]  = h0[i];
  c[i]  = c0[i];
  hb[i] = f2bf(h0[i]);
}

// ---------------- WMMA GEMMs ----------------
// One 16x16 tile per wave; used for Ukeys = enc_bf16 @ Ua^T + ub (M=4096,N=1024,K=1024)
__global__ void k_gemm_tiles(const unsigned short* __restrict__ A,
                             const unsigned short* __restrict__ Wp,
                             const float* __restrict__ bias,
                             float* __restrict__ C, int Mt, int N, int K) {
  int wid  = (blockIdx.x * blockDim.x + threadIdx.x) >> 5;
  int lane = threadIdx.x & 31;
  int mt = wid % Mt;
  int nt = wid / Mt;
  if (nt * 16 >= N) return;                 // wave-uniform
  v8f acc = {};
  int kt = K >> 5;
  size_t tb = (size_t)nt * kt;
  for (int kk = 0; kk < kt; ++kk) {
    v16bf bf = load_bfrag(Wp, tb + kk, lane);
    v16bf a  = load_afrag(A, mt * 16, K, kk << 5, lane);
    acc = __builtin_amdgcn_wmma_f32_16x16x32_bf16(false, a, false, bf, (short)0, acc,
                                                  false, false);
  }
  int n  = nt * 16 + (lane & 15);
  float bv = bias ? bias[n] : 0.f;
  int mh = (lane >> 4) << 3;
  #pragma unroll
  for (int r = 0; r < 8; ++r)
    C[(size_t)(mt * 16 + r + mh) * N + n] = acc[r] + bv;
}

// M=32 fixed (both 16-row tiles per wave, sharing B fragments). Optional fused
// second GEMM (A2 @ W2) and two biases. C is fp32, row stride ldc.
__global__ void k_gemm_step(const unsigned short* __restrict__ A1,
                            const unsigned short* __restrict__ W1, int K1,
                            const unsigned short* __restrict__ A2,
                            const unsigned short* __restrict__ W2, int K2,
                            const float* __restrict__ b1, const float* __restrict__ b2,
                            float* __restrict__ C, long long ldc, int N) {
  int wid  = (blockIdx.x * blockDim.x + threadIdx.x) >> 5;
  int lane = threadIdx.x & 31;
  int nt = wid;
  if (nt * 16 >= N) return;                 // wave-uniform
  v8f acc0 = {};
  v8f acc1 = {};
  {
    int kt = K1 >> 5;
    size_t tb = (size_t)nt * kt;
    for (int kk = 0; kk < kt; ++kk) {
      if (kk + 2 < kt)
        __builtin_prefetch((const void*)(W1 + (tb + kk + 2) * 512 + (size_t)lane * 16), 0, 1);
      v16bf bf = load_bfrag(W1, tb + kk, lane);
      v16bf a0 = load_afrag(A1, 0,  K1, kk << 5, lane);
      v16bf a1 = load_afrag(A1, 16, K1, kk << 5, lane);
      acc0 = __builtin_amdgcn_wmma_f32_16x16x32_bf16(false, a0, false, bf, (short)0, acc0,
                                                     false, false);
      acc1 = __builtin_amdgcn_wmma_f32_16x16x32_bf16(false, a1, false, bf, (short)0, acc1,
                                                     false, false);
    }
  }
  if (A2) {
    int kt = K2 >> 5;
    size_t tb = (size_t)nt * kt;
    for (int kk = 0; kk < kt; ++kk) {
      if (kk + 2 < kt)
        __builtin_prefetch((const void*)(W2 + (tb + kk + 2) * 512 + (size_t)lane * 16), 0, 1);
      v16bf bf = load_bfrag(W2, tb + kk, lane);
      v16bf a0 = load_afrag(A2, 0,  K2, kk << 5, lane);
      v16bf a1 = load_afrag(A2, 16, K2, kk << 5, lane);
      acc0 = __builtin_amdgcn_wmma_f32_16x16x32_bf16(false, a0, false, bf, (short)0, acc0,
                                                     false, false);
      acc1 = __builtin_amdgcn_wmma_f32_16x16x32_bf16(false, a1, false, bf, (short)0, acc1,
                                                     false, false);
    }
  }
  int n  = nt * 16 + (lane & 15);
  float bv = b1 ? b1[n] : 0.f;
  if (b2) bv += b2[n];
  int mh = (lane >> 4) << 3;
  #pragma unroll
  for (int r = 0; r < 8; ++r) {
    C[(size_t)(r + mh) * ldc + n]      = acc0[r] + bv;
    C[(size_t)(r + mh + 16) * ldc + n] = acc1[r] + bv;
  }
}

// ---------------- step kernels ----------------
// Embedding gather + LayerNorm -> xb[:, 0:H] (bf16). One block per batch row.
__global__ void k_embed_ln(const int* __restrict__ target, const float* __restrict__ emb,
                           const float* __restrict__ g, const float* __restrict__ be,
                           unsigned short* __restrict__ xb, int t) {
  __shared__ float red[256];
  int b = blockIdx.x, tid = threadIdx.x;
  int tok = (t == 0) ? 0 : target[b * TT + (t - 1)];
  const float* row = emb + (size_t)tok * HH;
  float s = 0.f;
  for (int h = tid; h < HH; h += 256) s += row[h];
  float mean = block_sum(s, red) * (1.f / HH);
  float v = 0.f;
  for (int h = tid; h < HH; h += 256) { float d = row[h] - mean; v += d * d; }
  float rstd = rsqrtf(block_sum(v, red) * (1.f / HH) + 1e-5f);
  for (int h = tid; h < HH; h += 256)
    xb[(size_t)b * (2 * HH) + h] = f2bf((row[h] - mean) * rstd * g[h] + be[h]);
}

// Bahdanau attention: scores -> softmax -> ctx -> LN(ctx) -> xb[:, H:2H]; w -> d_out.
__global__ void k_attn(const float* __restrict__ qW, const float* __restrict__ Ukeys,
                       const float* __restrict__ Va, const float* __restrict__ vb,
                       const float* __restrict__ temp, const int* __restrict__ vlen,
                       const float* __restrict__ enc, const float* __restrict__ cg,
                       const float* __restrict__ cb, unsigned short* __restrict__ xb,
                       float* __restrict__ attn_out, int t) {
  __shared__ float qsh[HH];
  __shared__ float sc[SS];
  __shared__ float red[256];
  int b = blockIdx.x, tid = threadIdx.x, lane = tid & 31, w = tid >> 5;
  for (int h = tid; h < HH; h += 256) qsh[h] = qW[(size_t)b * HH + h];
  __syncthreads();
  int vl = vlen[b];
  float invT = 1.f / temp[0];
  float vbb  = vb[0];
  for (int s = w; s < SS; s += 8) {
    const float* uk = Ukeys + ((size_t)b * SS + s) * HH;
    float sum = 0.f;
    for (int h = lane; h < HH; h += 32) sum += Va[h] * tanhf(qsh[h] + uk[h]);
    for (int m = 16; m > 0; m >>= 1) sum += __shfl_xor(sum, m, 32);
    if (lane == 0) sc[s] = (s < vl) ? (sum + vbb) * invT : -1e9f;
  }
  __syncthreads();
  // softmax over S = 128
  float x = (tid < SS) ? sc[tid] : -3.0e38f;
  red[tid] = x; __syncthreads();
  for (int o = 128; o > 0; o >>= 1) { if (tid < o) red[tid] = fmaxf(red[tid], red[tid + o]); __syncthreads(); }
  float mx = red[0]; __syncthreads();
  float ex = (tid < SS) ? __expf(x - mx) : 0.f;
  float den = block_sum(ex, red);
  if (tid < SS) {
    float wv = ex / den;
    sc[tid] = wv;
    attn_out[((size_t)b * TT + t) * SS + tid] = wv;
  }
  __syncthreads();
  // ctx = sum_s w[s] * enc[b, s, :], 4 h-values per thread
  float acc[4] = {0.f, 0.f, 0.f, 0.f};
  for (int s = 0; s < SS; ++s) {
    float wv = sc[s];
    const float* er = enc + ((size_t)b * SS + s) * HH;
    #pragma unroll
    for (int j = 0; j < 4; ++j) acc[j] += wv * er[tid + j * 256];
  }
  float s1 = acc[0] + acc[1] + acc[2] + acc[3];
  float mean = block_sum(s1, red) * (1.f / HH);
  float v2 = 0.f;
  #pragma unroll
  for (int j = 0; j < 4; ++j) { float d = acc[j] - mean; v2 += d * d; }
  float rstd = rsqrtf(block_sum(v2, red) * (1.f / HH) + 1e-5f);
  #pragma unroll
  for (int j = 0; j < 4; ++j) {
    int h = tid + j * 256;
    xb[(size_t)b * (2 * HH) + HH + h] = f2bf((acc[j] - mean) * rstd * cg[h] + cb[h]);
  }
}

// LSTM gate nonlinearity + state update for layer l. gates is (B, 4H) f32, order i,f,g,o.
__global__ void k_lstm_elem(const float* __restrict__ gates, float* __restrict__ h,
                            float* __restrict__ c, unsigned short* __restrict__ hb, int l) {
  int idx = blockIdx.x * 256 + threadIdx.x;   // grid covers BH exactly
  int b  = idx >> 10;
  int hh = idx & 1023;
  const float* g = gates + (size_t)b * (4 * HH);
  float ig = sigf(g[hh]);
  float fg = sigf(g[HH + hh]);
  float gg = tanhf(g[2 * HH + hh]);
  float og = sigf(g[3 * HH + hh]);
  size_t off = (size_t)l * BH + idx;
  float cn = fg * c[off] + ig * gg;
  float hn = og * tanhf(cn);
  c[off]  = cn;
  h[off]  = hn;
  hb[off] = f2bf(hn);
}

// LayerNorm of one (B,H) fp32 matrix -> bf16 output. One block per batch row.
__global__ void k_row_ln(const float* __restrict__ in, const float* __restrict__ g,
                         const float* __restrict__ be, unsigned short* __restrict__ out) {
  __shared__ float red[256];
  int b = blockIdx.x, tid = threadIdx.x;
  const float* row = in + (size_t)b * HH;
  float s = 0.f;
  for (int h = tid; h < HH; h += 256) s += row[h];
  float mean = block_sum(s, red) * (1.f / HH);
  float v = 0.f;
  for (int h = tid; h < HH; h += 256) { float d = row[h] - mean; v += d * d; }
  float rstd = rsqrtf(block_sum(v, red) * (1.f / HH) + 1e-5f);
  for (int h = tid; h < HH; h += 256)
    out[(size_t)b * HH + h] = f2bf((row[h] - mean) * rstd * g[h] + be[h]);
}

extern "C" void kernel_launch(void* const* d_in, const int* in_sizes, int n_in,
                              void* d_out, int out_size, void* d_ws, size_t ws_size,
                              hipStream_t stream) {
  (void)in_sizes; (void)n_in; (void)out_size; (void)ws_size;
  const float* enc    = (const float*)d_in[0];
  const float* h0     = (const float*)d_in[1];
  const float* c0     = (const float*)d_in[2];
  const int*   vlen   = (const int*)d_in[3];
  const int*   target = (const int*)d_in[4];
  const float* emb    = (const float*)d_in[5];
  const float* Wa     = (const float*)d_in[6];
  const float* ba     = (const float*)d_in[7];
  const float* Ua     = (const float*)d_in[8];
  const float* ub     = (const float*)d_in[9];
  const float* Va     = (const float*)d_in[10];
  const float* vb     = (const float*)d_in[11];
  const float* temp   = (const float*)d_in[12];
  const float* Wih[3] = {(const float*)d_in[13], (const float*)d_in[17], (const float*)d_in[21]};
  const float* Whh[3] = {(const float*)d_in[14], (const float*)d_in[18], (const float*)d_in[22]};
  const float* bih[3] = {(const float*)d_in[15], (const float*)d_in[19], (const float*)d_in[23]};
  const float* bhh[3] = {(const float*)d_in[16], (const float*)d_in[20], (const float*)d_in[24]};
  const float* fcw    = (const float*)d_in[25];
  const float* fcb    = (const float*)d_in[26];
  const float* eng    = (const float*)d_in[27];
  const float* enb    = (const float*)d_in[28];
  const float* cng    = (const float*)d_in[29];
  const float* cnb    = (const float*)d_in[30];
  const float* ong    = (const float*)d_in[31];
  const float* onb    = (const float*)d_in[32];

  // ---- workspace carve-out (256B aligned blocks) ----
  char* base = (char*)d_ws;
  size_t off = 0;
  auto alloc = [&](size_t bytes) -> char* {
    char* p = base + off;
    off = (off + bytes + 255) & ~(size_t)255;
    return p;
  };
  unsigned short* enc_bf = (unsigned short*)alloc((size_t)BSH * 2);
  float*          Ukeys  = (float*)alloc((size_t)BSH * 4);
  unsigned short* Wa_p   = (unsigned short*)alloc((size_t)HH * HH * 2);
  unsigned short* Ua_p   = (unsigned short*)alloc((size_t)HH * HH * 2);
  unsigned short* Wih_p[3];
  unsigned short* Whh_p[3];
  Wih_p[0] = (unsigned short*)alloc((size_t)4 * HH * 2 * HH * 2);
  Wih_p[1] = (unsigned short*)alloc((size_t)4 * HH * HH * 2);
  Wih_p[2] = (unsigned short*)alloc((size_t)4 * HH * HH * 2);
  Whh_p[0] = (unsigned short*)alloc((size_t)4 * HH * HH * 2);
  Whh_p[1] = (unsigned short*)alloc((size_t)4 * HH * HH * 2);
  Whh_p[2] = (unsigned short*)alloc((size_t)4 * HH * HH * 2);
  unsigned short* fcw_p = (unsigned short*)alloc((size_t)VV * HH * 2);
  float*          hst   = (float*)alloc((size_t)LL * BH * 4);
  float*          cst   = (float*)alloc((size_t)LL * BH * 4);
  unsigned short* hb    = (unsigned short*)alloc((size_t)LL * BH * 2);
  unsigned short* xb    = (unsigned short*)alloc((size_t)BB * 2 * HH * 2);
  float*          qW    = (float*)alloc((size_t)BH * 4);
  float*          gates = (float*)alloc((size_t)BB * 4 * HH * 4);
  unsigned short* outb  = (unsigned short*)alloc((size_t)BH * 2);

  // ---- one-time prep: bf16 conversion + WMMA fragment packing ----
  k_pack_w<<<1024, 256, 0, stream>>>(Wa, Wa_p, HH, HH);
  k_pack_w<<<1024, 256, 0, stream>>>(Ua, Ua_p, HH, HH);
  k_pack_w<<<2048, 256, 0, stream>>>(Wih[0], Wih_p[0], 4 * HH, 2 * HH);
  k_pack_w<<<2048, 256, 0, stream>>>(Wih[1], Wih_p[1], 4 * HH, HH);
  k_pack_w<<<2048, 256, 0, stream>>>(Wih[2], Wih_p[2], 4 * HH, HH);
  k_pack_w<<<2048, 256, 0, stream>>>(Whh[0], Whh_p[0], 4 * HH, HH);
  k_pack_w<<<2048, 256, 0, stream>>>(Whh[1], Whh_p[1], 4 * HH, HH);
  k_pack_w<<<2048, 256, 0, stream>>>(Whh[2], Whh_p[2], 4 * HH, HH);
  k_pack_w<<<4096, 256, 0, stream>>>(fcw, fcw_p, VV, HH);
  k_f2bf<<<2048, 256, 0, stream>>>(enc, enc_bf, (size_t)BSH);
  k_init_state<<<(LL * BH) / 256, 256, 0, stream>>>(h0, c0, hst, cst, hb);

  // Ukeys = enc @ Ua^T + ub   (M=4096, N=1024, K=1024) -> 16384 waves
  k_gemm_tiles<<<2048, 256, 0, stream>>>(enc_bf, Ua_p, ub, Ukeys, (BB * SS) / 16, HH, HH);

  float* logits   = (float*)d_out;                                  // (B,T,V)
  float* hT_out   = (float*)d_out + (size_t)BB * TT * VV;           // (L,B,H)
  float* cT_out   = hT_out + (size_t)LL * BH;                       // (L,B,H)
  float* attn_out = cT_out + (size_t)LL * BH;                       // (B,T,S)

  for (int t = 0; t < TT; ++t) {
    k_embed_ln<<<BB, 256, 0, stream>>>(target, emb, eng, enb, xb, t);
    // qW = h_top @ Wa^T + ba  (N=1024 -> 64 waves -> 8 blocks)
    k_gemm_step<<<8, 256, 0, stream>>>(hb + 2 * BH, Wa_p, HH, nullptr, nullptr, 0,
                                       ba, nullptr, qW, (long long)HH, HH);
    k_attn<<<BB, 256, 0, stream>>>(qW, Ukeys, Va, vb, temp, vlen, enc, cng, cnb,
                                   xb, attn_out, t);
    // layer 0: gates = x @ Wih0^T + bih0 + h0 @ Whh0^T + bhh0   (N=4096 -> 32 blocks)
    k_gemm_step<<<32, 256, 0, stream>>>(xb, Wih_p[0], 2 * HH, hb, Whh_p[0], HH,
                                        bih[0], bhh[0], gates, (long long)(4 * HH), 4 * HH);
    k_lstm_elem<<<BH / 256, 256, 0, stream>>>(gates, hst, cst, hb, 0);
    // layer 1
    k_gemm_step<<<32, 256, 0, stream>>>(hb, Wih_p[1], HH, hb + BH, Whh_p[1], HH,
                                        bih[1], bhh[1], gates, (long long)(4 * HH), 4 * HH);
    k_lstm_elem<<<BH / 256, 256, 0, stream>>>(gates, hst, cst, hb, 1);
    // layer 2
    k_gemm_step<<<32, 256, 0, stream>>>(hb + BH, Wih_p[2], HH, hb + 2 * BH, Whh_p[2], HH,
                                        bih[2], bhh[2], gates, (long long)(4 * HH), 4 * HH);
    k_lstm_elem<<<BH / 256, 256, 0, stream>>>(gates, hst, cst, hb, 2);
    // out = LN(h2) -> bf16
    k_row_ln<<<BB, 256, 0, stream>>>(hst + 2 * BH, ong, onb, outb);
    // logits[:, t, :] = out @ fc_w^T + fc_b  (N=32000 -> 2000 waves -> 250 blocks)
    k_gemm_step<<<250, 256, 0, stream>>>(outb, fcw_p, HH, nullptr, nullptr, 0,
                                         fcb, nullptr, logits + (size_t)t * VV,
                                         (long long)TT * VV, VV);
  }

  hipMemcpyAsync(hT_out, hst, (size_t)LL * BH * sizeof(float),
                 hipMemcpyDeviceToDevice, stream);
  hipMemcpyAsync(cT_out, cst, (size_t)LL * BH * sizeof(float),
                 hipMemcpyDeviceToDevice, stream);
}
